// FusionMamba_20066087207631
// MI455X (gfx1250) — compile-verified, matrix operand order checked by reading, not run on hardware
//
#include <hip/hip_runtime.h>
#include <hip/hip_bf16.h>

// ---------------- problem constants ----------------
#define DD    64          // feature dim
#define LSEQ  16384       // sequence length (128*128)
#define BATCH 4
#define NST   8           // SSM states
#define RLOW  4           // dt low rank
#define CHUNK 256         // scan chunk length
#define NCHUNK (LSEQ / CHUNK)        // 64
#define BL    (BATCH * (long)LSEQ)   // 65536 tokens
#define BLD   (BL * DD)              // 4,194,304 elems

typedef __attribute__((ext_vector_type(16))) __bf16 v16bf;
typedef __attribute__((ext_vector_type(8)))  __bf16 v8bf;
typedef __attribute__((ext_vector_type(2)))  __bf16 v2bf;
typedef __attribute__((ext_vector_type(8)))  float  v8f;

// ---------------- WMMA helpers (CDNA5 16x16x32 bf16, wave32) ----------------
// ISA 7.12.2 16-bit operand layout: for lane half hi = lane>>4,
//   frag[j], j in [0,8)  -> K = kb + hi*8 + j
//   frag[j], j in [8,16) -> K = kb + 16 + hi*8 + (j-8)
// i.e. two contiguous 8-element (16B) runs -> two ds_load_b128 per fragment,
// provided the operand is stored K-contiguous per row (A: tokens x K;
// B: weights staged TRANSPOSED as N x K).
__device__ __forceinline__ v16bf load_row_frag(const __bf16* row, int kb, int hi) {
  v8bf lo = *(const v8bf*)(row + kb + hi * 8);
  v8bf hh = *(const v8bf*)(row + kb + 16 + hi * 8);
  return __builtin_shufflevector(lo, hh, 0, 1, 2, 3, 4, 5, 6, 7,
                                 8, 9, 10, 11, 12, 13, 14, 15);
}

__device__ __forceinline__ v8f wmma_bf16(v16bf a, v16bf b, v8f c) {
  return __builtin_amdgcn_wmma_f32_16x16x32_bf16(
      /*neg_a=*/false, a, /*neg_b=*/false, b,
      /*c_mod=*/(short)0, c, /*reuse_a=*/false, /*reuse_b=*/false);
}

__device__ __forceinline__ float siluf(float x) { return x / (1.f + __expf(-x)); }
__device__ __forceinline__ float softplusf(float x) {
  return (x > 20.f) ? x : log1pf(__expf(x));
}

// ================= P1: LN0 -> W_in(x,z); LN1 -> W_ine -> W_x -> dt/B/C ======
__global__ __launch_bounds__(256) void k_pre(
    const float* __restrict__ X0, long sb0, long sl0, long sd0,
    const float* __restrict__ X1, long sb1, long sl1, long sd1,
    const float* __restrict__ g0v, const float* __restrict__ b0v,
    const float* __restrict__ g1v, const float* __restrict__ b1v,
    const float* __restrict__ Win, const float* __restrict__ Wine,
    const float* __restrict__ Wx,  const float* __restrict__ Wdt,
    const float* __restrict__ bdtp,
    float* __restrict__ xout, float* __restrict__ zout,
    float* __restrict__ dt0, float* __restrict__ dt1,
    float* __restrict__ Bm0, float* __restrict__ Cm0,
    float* __restrict__ Bm1, float* __restrict__ Cm1) {
  // all weight tiles staged TRANSPOSED: row n holds K=0..63 contiguous
  __shared__ __align__(16) __bf16 sWinT[128 * DD];
  __shared__ __align__(16) __bf16 sWineT[DD * DD];
  __shared__ __align__(16) __bf16 sWxT[2][32 * DD];   // cols >=20 zero-padded
  __shared__ float  sWdt[2][RLOW * DD];
  __shared__ float  sbdt[2][DD];
  __shared__ float  sg0[DD], sB0[DD], sg1[DD], sB1[DD];
  __shared__ __align__(16) __bf16 tileA[8][16 * DD];
  __shared__ float  tileDtr[8][16 * RLOW];

  int tid = threadIdx.x;
  for (int idx = tid; idx < 128 * DD; idx += 256) {
    int n = idx >> 6, k = idx & (DD - 1);
    sWinT[idx] = (__bf16)Win[k * 128 + n];
  }
  for (int idx = tid; idx < DD * DD; idx += 256) {
    int n = idx >> 6, k = idx & (DD - 1);
    sWineT[idx] = (__bf16)Wine[k * DD + n];
  }
  for (int idx = tid; idx < 2 * 32 * DD; idx += 256) {
    int d2 = idx >> 11, rem = idx & 2047;
    int n = rem >> 6, k = rem & (DD - 1);
    (&sWxT[0][0])[idx] = (n < 20) ? (__bf16)Wx[(d2 * DD + k) * 20 + n]
                                  : (__bf16)0.f;
  }
  for (int idx = tid; idx < 2 * RLOW * DD; idx += 256) (&sWdt[0][0])[idx] = Wdt[idx];
  for (int idx = tid; idx < 2 * DD; idx += 256) (&sbdt[0][0])[idx] = bdtp[idx];
  if (tid < DD) { sg0[tid] = g0v[tid]; sB0[tid] = b0v[tid];
                  sg1[tid] = g1v[tid]; sB1[tid] = b1v[tid]; }
  __syncthreads();

  int wave = tid >> 5, lane = tid & 31;
  int t = lane & 15, hf = lane >> 4;
  long tile = (long)blockIdx.x * 8 + wave;
  long tok0 = tile * 16;
  long tok  = tok0 + t;
  long bb = tok >> 14, ll = tok & (LSEQ - 1);
  __bf16* myt = &tileA[wave][0];
  const __bf16* arow = myt + (lane & 15) * DD;            // A row for this lane
  int bcol = lane & 15;                                    // B col for this lane

  // ---- LayerNorm of X0 into bf16 A-tile ----
  {
    const float* xp = X0 + bb * sb0 + ll * sl0;
    __builtin_prefetch(xp, 0, 1);
    float vals[32], s = 0.f, s2 = 0.f;
#pragma unroll
    for (int k = 0; k < 32; ++k) {
      float v = xp[(hf * 32 + k) * sd0];
      vals[k] = v; s += v; s2 += v * v;
    }
    s  += __shfl_xor(s, 16, 32);
    s2 += __shfl_xor(s2, 16, 32);
    float mean = s * (1.f / DD);
    float inv  = rsqrtf(s2 * (1.f / DD) - mean * mean + 1e-5f);
#pragma unroll
    for (int kk = 0; kk < 16; ++kk) {
      int dd = hf * 32 + 2 * kk;
      v2bf p;
      p[0] = (__bf16)((vals[2 * kk]     - mean) * inv * sg0[dd]     + sB0[dd]);
      p[1] = (__bf16)((vals[2 * kk + 1] - mean) * inv * sg0[dd + 1] + sB0[dd + 1]);
      *(v2bf*)(myt + t * DD + dd) = p;
    }
  }
  __syncthreads();

  // ---- xz = ln(a) @ W_in (64 -> 128), split into x / z ----
  {
    v16bf a0 = load_row_frag(arow, 0, hf);
    v16bf a1 = load_row_frag(arow, 32, hf);
#pragma unroll
    for (int nt = 0; nt < 8; ++nt) {
      const __bf16* brow = sWinT + (nt * 16 + bcol) * DD;
      v8f acc = {};
      acc = wmma_bf16(a0, load_row_frag(brow, 0, hf), acc);
      acc = wmma_bf16(a1, load_row_frag(brow, 32, hf), acc);
      int c = nt * 16 + bcol;
      float* dst = (c < DD) ? xout : zout;
      int cc = c & (DD - 1);
#pragma unroll
      for (int r = 0; r < 8; ++r) {
        long tk = tok0 + hf * 8 + r;
        dst[tk * DD + cc] = acc[r];
      }
    }
  }
  __syncthreads();

  // ---- LayerNorm of X1 into the same tile ----
  {
    const float* xp = X1 + bb * sb1 + ll * sl1;
    __builtin_prefetch(xp, 0, 1);
    float vals[32], s = 0.f, s2 = 0.f;
#pragma unroll
    for (int k = 0; k < 32; ++k) {
      float v = xp[(hf * 32 + k) * sd1];
      vals[k] = v; s += v; s2 += v * v;
    }
    s  += __shfl_xor(s, 16, 32);
    s2 += __shfl_xor(s2, 16, 32);
    float mean = s * (1.f / DD);
    float inv  = rsqrtf(s2 * (1.f / DD) - mean * mean + 1e-5f);
#pragma unroll
    for (int kk = 0; kk < 16; ++kk) {
      int dd = hf * 32 + 2 * kk;
      v2bf p;
      p[0] = (__bf16)((vals[2 * kk]     - mean) * inv * sg1[dd]     + sB1[dd]);
      p[1] = (__bf16)((vals[2 * kk + 1] - mean) * inv * sg1[dd + 1] + sB1[dd + 1]);
      *(v2bf*)(myt + t * DD + dd) = p;
    }
  }
  __syncthreads();

  // ---- ee = ln(e) @ W_ine (64 -> 64), kept in regs, re-staged as bf16 ----
  float eev[4][8];
  {
    v16bf a0 = load_row_frag(arow, 0, hf);
    v16bf a1 = load_row_frag(arow, 32, hf);
#pragma unroll
    for (int nt = 0; nt < 4; ++nt) {
      const __bf16* brow = sWineT + (nt * 16 + bcol) * DD;
      v8f acc = {};
      acc = wmma_bf16(a0, load_row_frag(brow, 0, hf), acc);
      acc = wmma_bf16(a1, load_row_frag(brow, 32, hf), acc);
#pragma unroll
      for (int r = 0; r < 8; ++r) eev[nt][r] = acc[r];
    }
  }
  __syncthreads();
#pragma unroll
  for (int nt = 0; nt < 4; ++nt)
#pragma unroll
    for (int r = 0; r < 8; ++r)
      myt[(hf * 8 + r) * DD + nt * 16 + bcol] = (__bf16)eev[nt][r];
  __syncthreads();

  // ---- dbl = ee @ W_x[d] (64 -> 20) -> dtr | Bm | Cm ; dt = softplus(...) ----
  v16bf a0 = load_row_frag(arow, 0, hf);
  v16bf a1 = load_row_frag(arow, 32, hf);
  for (int d2 = 0; d2 < 2; ++d2) {
    float* Bmp = d2 ? Bm1 : Bm0;
    float* Cmp = d2 ? Cm1 : Cm0;
#pragma unroll
    for (int nt = 0; nt < 2; ++nt) {
      const __bf16* brow = &sWxT[d2][0] + (nt * 16 + bcol) * DD;
      v8f acc = {};
      acc = wmma_bf16(a0, load_row_frag(brow, 0, hf), acc);
      acc = wmma_bf16(a1, load_row_frag(brow, 32, hf), acc);
      int c = nt * 16 + bcol;
#pragma unroll
      for (int r = 0; r < 8; ++r) {
        long tk = tok0 + hf * 8 + r;
        if (c < RLOW)          tileDtr[wave][(hf * 8 + r) * RLOW + c] = acc[r];
        else if (c < RLOW + 8) Bmp[tk * NST + (c - RLOW)]     = acc[r];
        else if (c < 20)       Cmp[tk * NST + (c - RLOW - 8)] = acc[r];
      }
    }
    __syncthreads();
    float* dtp = d2 ? dt1 : dt0;
#pragma unroll
    for (int k = 0; k < 32; ++k) {
      int dd = hf * 32 + k;
      float sdt = sbdt[d2][dd];
#pragma unroll
      for (int r2 = 0; r2 < RLOW; ++r2)
        sdt += tileDtr[wave][t * RLOW + r2] * sWdt[d2][r2 * DD + dd];
      dtp[tok * DD + dd] = softplusf(sdt);
    }
    __syncthreads();
  }
}

// ================= P2: depthwise causal (d=0) + anticausal (d=1) conv ======
__global__ __launch_bounds__(256) void k_conv(
    const float* __restrict__ x, const float* __restrict__ cW,
    const float* __restrict__ cB, float* __restrict__ xc0,
    float* __restrict__ xc1) {
  long idx = (long)blockIdx.x * 256 + threadIdx.x;  // BLD threads
  int dd = idx & (DD - 1);
  long tok = idx >> 6;
  long ll = tok & (LSEQ - 1);
  float tap[7];
#pragma unroll
  for (int o = -3; o <= 3; ++o) {
    long l2 = ll + o;
    tap[o + 3] = (l2 >= 0 && l2 < LSEQ) ? x[(tok + o) * DD + dd] : 0.f;
  }
  float a0 = cB[dd], a1 = cB[DD + dd];
#pragma unroll
  for (int k = 0; k < 4; ++k) {
    a0 += cW[dd * 4 + k] * tap[k];                 // x[l-3+k]
    a1 += cW[256 + dd * 4 + k] * tap[6 - k];       // x[l+3-k]
  }
  xc0[idx] = siluf(a0);
  xc1[idx] = siluf(a1);
}

// ====== P3: chunked scan phase 1 — per-chunk decay product P, local end S ===
__global__ __launch_bounds__(512) void k_scan1(
    const float* __restrict__ dt0, const float* __restrict__ dt1,
    const float* __restrict__ xc0, const float* __restrict__ xc1,
    const float* __restrict__ Bm0, const float* __restrict__ Bm1,
    const float* __restrict__ Al, float* __restrict__ P,
    float* __restrict__ S) {
  int chunk = blockIdx.x, b = blockIdx.y, dir = blockIdx.z;
  int tid = threadIdx.x;           // d*8+n
  int d = tid >> 3, n = tid & 7;
  const float* dt = dir ? dt1 : dt0;
  const float* xc = dir ? xc1 : xc0;
  const float* Bm = dir ? Bm1 : Bm0;
  float Aa = -__expf(Al[dir * DD * NST + d * NST + n]);
  float h = 0.f, Pp = 1.f;
  long base = (long)b * LSEQ;
  for (int t2 = 0; t2 < CHUNK; ++t2) {
    long p = (long)chunk * CHUNK + t2;
    long l = dir ? (LSEQ - 1 - p) : p;
    long g = (base + l) * DD;
    float dtv = dt[g + d];
    float xv  = xc[g + d];
    float Bv  = Bm[(base + l) * NST + n];
    float dA  = __expf(dtv * Aa);
    h = h * dA + dtv * xv * Bv;
    Pp *= dA;
  }
  long o = (((long)dir * BATCH + b) * NCHUNK + chunk) * 512 + tid;
  P[o] = Pp;
  S[o] = h;
}

// ====== P4: chunk prefix — initial state H0 for every chunk =================
__global__ __launch_bounds__(512) void k_scan2(
    const float* __restrict__ P, const float* __restrict__ S,
    float* __restrict__ H0) {
  long sbi = blockIdx.x;  // dir*BATCH + b, grid 8
  int tid = threadIdx.x;
  float H = 0.f;
  for (int j = 0; j < NCHUNK; ++j) {
    long o = (sbi * NCHUNK + j) * 512 + tid;
    H0[o] = H;
    H = P[o] * H + S[o];
  }
}

// ====== P5: phase 3 — recompute with correct init, emit y*silu(z) ===========
__global__ __launch_bounds__(64) void k_scan3(
    const float* __restrict__ dt0, const float* __restrict__ dt1,
    const float* __restrict__ xc0, const float* __restrict__ xc1,
    const float* __restrict__ Bm0, const float* __restrict__ Bm1,
    const float* __restrict__ Cm0, const float* __restrict__ Cm1,
    const float* __restrict__ Al, const float* __restrict__ Dpv,
    const float* __restrict__ zb, const float* __restrict__ H0,
    float* __restrict__ yb0, float* __restrict__ yb1) {
  int chunk = blockIdx.x, b = blockIdx.y, dir = blockIdx.z;
  int d = threadIdx.x;  // 64
  const float* dt = dir ? dt1 : dt0;
  const float* xc = dir ? xc1 : xc0;
  const float* Bm = dir ? Bm1 : Bm0;
  const float* Cm = dir ? Cm1 : Cm0;
  float* yb = dir ? yb1 : yb0;
  float A[NST], h[NST];
  long ho = (((long)dir * BATCH + b) * NCHUNK + chunk) * 512 + d * NST;
#pragma unroll
  for (int n = 0; n < NST; ++n) {
    A[n] = -__expf(Al[dir * DD * NST + d * NST + n]);
    h[n] = H0[ho + n];
  }
  float Dv = Dpv[dir * DD + d];
  __shared__ float sB[16][NST];
  __shared__ float sC[16][NST];
  long base = (long)b * LSEQ;
  for (int t0 = 0; t0 < CHUNK; t0 += 16) {
    __syncthreads();
#pragma unroll
    for (int q = 0; q < 4; ++q) {
      int j = d * 4 + q;       // 0..255 = 16 steps x 16 values
      int st = j >> 4, c = j & 15;
      long p = (long)chunk * CHUNK + t0 + st;
      long l = dir ? (LSEQ - 1 - p) : p;
      if (c < NST) sB[st][c] = Bm[(base + l) * NST + c];
      else         sC[st][c - NST] = Cm[(base + l) * NST + (c - NST)];
    }
    __syncthreads();
    for (int tt = 0; tt < 16; ++tt) {
      long p = (long)chunk * CHUNK + t0 + tt;
      long l = dir ? (LSEQ - 1 - p) : p;
      long g = (base + l) * DD + d;
      float dtv = dt[g], xv = xc[g];
      float y = 0.f;
#pragma unroll
      for (int n = 0; n < NST; ++n) {
        float dA = __expf(dtv * A[n]);
        h[n] = h[n] * dA + dtv * xv * sB[tt][n];
        y += h[n] * sC[tt][n];
      }
      float zv = zb[g];
      yb[g] = (y + xv * Dv) * siluf(zv);
    }
  }
}

// ================= P6: (y0+y1) @ W_out + b_out + skip =======================
__global__ __launch_bounds__(256) void k_out(
    const float* __restrict__ yb0, const float* __restrict__ yb1,
    const float* __restrict__ skipX, long sb, long sl, long sd,
    const float* __restrict__ Wout, const float* __restrict__ boutv,
    float* __restrict__ XO) {
  __shared__ __align__(16) __bf16 sWT[DD * DD];   // transposed
  __shared__ float  sBv[DD];
  __shared__ __align__(16) __bf16 tileA[8][16 * DD];
  int tid = threadIdx.x;
  for (int idx = tid; idx < DD * DD; idx += 256) {
    int n = idx >> 6, k = idx & (DD - 1);
    sWT[idx] = (__bf16)Wout[k * DD + n];
  }
  if (tid < DD) sBv[tid] = boutv[tid];
  __syncthreads();

  int wave = tid >> 5, lane = tid & 31;
  int t = lane & 15, hf = lane >> 4;
  int bcol = lane & 15;
  long tok0 = ((long)blockIdx.x * 8 + wave) * 16;
  __bf16* myt = &tileA[wave][0];
#pragma unroll
  for (int kk = 0; kk < 16; ++kk) {
    int dd = hf * 32 + 2 * kk;
    long g = (tok0 + t) * DD + dd;
    v2bf p;
    p[0] = (__bf16)(yb0[g] + yb1[g]);
    p[1] = (__bf16)(yb0[g + 1] + yb1[g + 1]);
    *(v2bf*)(myt + t * DD + dd) = p;
  }
  __syncthreads();
  const __bf16* arow = myt + bcol * DD;
  v16bf a0 = load_row_frag(arow, 0, hf);
  v16bf a1 = load_row_frag(arow, 32, hf);
#pragma unroll
  for (int nt = 0; nt < 4; ++nt) {
    const __bf16* brow = sWT + (nt * 16 + bcol) * DD;
    v8f acc = {};
    acc = wmma_bf16(a0, load_row_frag(brow, 0, hf), acc);
    acc = wmma_bf16(a1, load_row_frag(brow, 32, hf), acc);
    int c = nt * 16 + bcol;
#pragma unroll
    for (int r = 0; r < 8; ++r) {
      long tk = tok0 + hf * 8 + r;
      long bb = tk >> 14, ll = tk & (LSEQ - 1);
      XO[tk * DD + c] = acc[r] + sBv[c] + skipX[bb * sb + ll * sl + c * sd];
    }
  }
}

// ============ P7: fusion GEMM + residual + transpose to (B,D,H,W) ===========
__global__ __launch_bounds__(256) void k_fuse(
    const float* __restrict__ spa, const float* __restrict__ spe,
    const float* __restrict__ x1b, const float* __restrict__ x2b,
    const float* __restrict__ Wfo, const float* __restrict__ bfov,
    float* __restrict__ o0, float* __restrict__ o1) {
  __shared__ __align__(16) __bf16 sWT[DD * DD];   // transposed
  __shared__ float  sBv[DD];
  __shared__ __align__(16) __bf16 tileA[8][16 * DD];
  int tid = threadIdx.x;
  for (int idx = tid; idx < DD * DD; idx += 256) {
    int n = idx >> 6, k = idx & (DD - 1);
    sWT[idx] = (__bf16)Wfo[k * DD + n];
  }
  if (tid < DD) sBv[tid] = bfov[tid];
  __syncthreads();

  int wave = tid >> 5, lane = tid & 31;
  int t = lane & 15, hf = lane >> 4;
  int bcol = lane & 15;
  long tok0 = ((long)blockIdx.x * 8 + wave) * 16;
  __bf16* myt = &tileA[wave][0];
#pragma unroll
  for (int kk = 0; kk < 16; ++kk) {
    int dd = hf * 32 + 2 * kk;
    long g = (tok0 + t) * DD + dd;
    v2bf p;
    p[0] = (__bf16)(spa[g] + spe[g]);
    p[1] = (__bf16)(spa[g + 1] + spe[g + 1]);
    *(v2bf*)(myt + t * DD + dd) = p;
  }
  __syncthreads();
  const __bf16* arow = myt + bcol * DD;
  v16bf a0 = load_row_frag(arow, 0, hf);
  v16bf a1 = load_row_frag(arow, 32, hf);
#pragma unroll
  for (int nt = 0; nt < 4; ++nt) {
    const __bf16* brow = sWT + (nt * 16 + bcol) * DD;
    v8f acc = {};
    acc = wmma_bf16(a0, load_row_frag(brow, 0, hf), acc);
    acc = wmma_bf16(a1, load_row_frag(brow, 32, hf), acc);
    int c = nt * 16 + bcol;
#pragma unroll
    for (int r = 0; r < 8; ++r) {
      long tk = tok0 + hf * 8 + r;
      long bb = tk >> 14, ll = tk & (LSEQ - 1);
      float f = acc[r] + sBv[c];
      long oidx = (bb * DD + c) * (long)LSEQ + ll;   // (B,D,H,W)
      o0[oidx] = x1b[tk * DD + c] + f;
      o1[oidx] = x2b[tk * DD + c] + f;
    }
  }
}

// ============================ host orchestration ============================
extern "C" void kernel_launch(void* const* d_in, const int* in_sizes, int n_in,
                              void* d_out, int out_size, void* d_ws,
                              size_t ws_size, hipStream_t stream) {
  (void)in_sizes; (void)n_in; (void)out_size; (void)ws_size;
  const float* img1     = (const float*)d_in[0];
  const float* img2     = (const float*)d_in[1];
  const float* img1_sum = (const float*)d_in[2];
  const float* img2_sum = (const float*)d_in[3];
  const float* ln_g0 = (const float*)d_in[4];
  const float* ln_b0 = (const float*)d_in[5];
  const float* ln_g1 = (const float*)d_in[6];
  const float* ln_b1 = (const float*)d_in[7];
  const float* W_in  = (const float*)d_in[8];   // (4,64,128)
  const float* W_ine = (const float*)d_in[9];   // (4,64,64)
  const float* convW = (const float*)d_in[10];  // (4,2,64,4)
  const float* convB = (const float*)d_in[11];  // (4,2,64)
  const float* W_x   = (const float*)d_in[12];  // (4,2,64,20)
  const float* W_dt  = (const float*)d_in[13];  // (4,2,4,64)
  const float* b_dt  = (const float*)d_in[14];  // (4,2,64)
  const float* A_log = (const float*)d_in[15];  // (4,2,64,8)
  const float* Dp    = (const float*)d_in[16];  // (4,2,64)
  const float* W_out = (const float*)d_in[17];  // (4,64,64)
  const float* b_out = (const float*)d_in[18];  // (4,64)
  const float* W_fo  = (const float*)d_in[19];  // (64,64)
  const float* b_fo  = (const float*)d_in[20];  // (64,)
  float* out = (float*)d_out;

  float* ws = (float*)d_ws;
  const long BL8 = BL * NST;                 // 524288
  float* xbuf = ws + 0 * BLD;
  float* zbuf = ws + 1 * BLD;
  float* dt0  = ws + 2 * BLD;
  float* dt1  = ws + 3 * BLD;
  float* xc0  = ws + 4 * BLD;
  float* xc1  = ws + 5 * BLD;
  float* Bm0  = ws + 6 * BLD;
  float* Cm0  = Bm0 + BL8;
  float* Bm1  = Cm0 + BL8;
  float* Cm1  = Bm1 + BL8;
  float* Pb   = Cm1 + BL8;                   // 2*4*64*512 = 262144
  float* Sb   = Pb + 262144;
  float* H0b  = Sb + 262144;
  float* yb0  = ws + 7 * BLD;
  float* yb1  = ws + 8 * BLD;
  float* x1b  = ws + 9 * BLD;
  float* x2b  = ws + 10 * BLD;
  float* spa  = ws + 11 * BLD;
  float* spe  = ws + 12 * BLD;

  const long SB_IMG = (long)DD * LSEQ, SL_IMG = 1, SD_IMG = LSEQ;  // (B,D,H,W)
  const long SB_TOK = (long)LSEQ * DD, SL_TOK = DD, SD_TOK = 1;    // (B,L,D)

  dim3 gScan(NCHUNK, BATCH, 2);

  auto run_block = [&](int i, const float* X0, bool tok0layout,
                       const float* X1, bool tok1layout, float* XO) {
    long sb0 = tok0layout ? SB_TOK : SB_IMG, sl0 = tok0layout ? SL_TOK : SL_IMG,
         sd0 = tok0layout ? SD_TOK : SD_IMG;
    long sb1 = tok1layout ? SB_TOK : SB_IMG, sl1 = tok1layout ? SL_TOK : SL_IMG,
         sd1 = tok1layout ? SD_TOK : SD_IMG;
    k_pre<<<512, 256, 0, stream>>>(
        X0, sb0, sl0, sd0, X1, sb1, sl1, sd1,
        ln_g0 + i * DD, ln_b0 + i * DD, ln_g1 + i * DD, ln_b1 + i * DD,
        W_in + (long)i * DD * 128, W_ine + (long)i * DD * DD,
        W_x + (long)i * 2 * DD * 20, W_dt + (long)i * 2 * RLOW * DD,
        b_dt + (long)i * 2 * DD,
        xbuf, zbuf, dt0, dt1, Bm0, Cm0, Bm1, Cm1);
    k_conv<<<BLD / 256, 256, 0, stream>>>(
        xbuf, convW + (long)i * 2 * DD * 4, convB + (long)i * 2 * DD, xc0, xc1);
    k_scan1<<<gScan, 512, 0, stream>>>(dt0, dt1, xc0, xc1, Bm0, Bm1,
                                       A_log + (long)i * 2 * DD * NST, Pb, Sb);
    k_scan2<<<2 * BATCH, 512, 0, stream>>>(Pb, Sb, H0b);
    k_scan3<<<gScan, 64, 0, stream>>>(dt0, dt1, xc0, xc1, Bm0, Bm1, Cm0, Cm1,
                                      A_log + (long)i * 2 * DD * NST,
                                      Dp + (long)i * 2 * DD, zbuf, H0b, yb0, yb1);
    k_out<<<512, 256, 0, stream>>>(yb0, yb1, X0, sb0, sl0, sd0,
                                   W_out + (long)i * DD * DD, b_out + i * DD, XO);
  };

  run_block(0, img1, false, img1_sum, false, x1b);
  run_block(1, img2, false, img2_sum, false, x2b);
  run_block(2, x1b, true, x2b, true, spa);
  run_block(3, x2b, true, x1b, true, spe);

  k_fuse<<<512, 256, 0, stream>>>(spa, spe, x1b, x2b, W_fo, b_fo,
                                  out, out + BLD);
  hipMemcpyAsync(out + 2 * BLD, img1_sum, BLD * sizeof(float),
                 hipMemcpyDeviceToDevice, stream);
  hipMemcpyAsync(out + 3 * BLD, img2_sum, BLD * sizeof(float),
                 hipMemcpyDeviceToDevice, stream);
}
